// GPTQMarlinSparseMoELayer_70042326663708
// MI455X (gfx1250) — matrix operand order repaired and checked by reading, not compiled, not run on hardware
//
#include <hip/hip_runtime.h>
#include <stdint.h>

// ---- problem dims (fixed by harness setup_inputs) ----
#define T_TOK 2048
#define K_DIM 1024
#define E_EXP 8
#define N_DIM 2816
#define N2    (2 * N_DIM)     // 5632
#define GRP   128
#define TOPK  2
#define MT    64              // rows per tile (GEMM1)
#define MT2   128             // rows per tile (GEMM2)
#define CT1   64              // output cols per block (GEMM1, weight-resident)
#define WP    513             // LDS weight row stride (pairs) -> conflict-free banks

typedef __attribute__((ext_vector_type(16))) __bf16 v16bf;
typedef __attribute__((ext_vector_type(8)))  float  v8f;
typedef __attribute__((address_space(3))) uint32_t lds_u32;

union Frag { v16bf v; uint32_t u[8]; };

// float -> bf16 round-to-nearest-even (pure integer)
static __device__ __forceinline__ uint16_t f2bf(float f) {
  uint32_t u = __builtin_bit_cast(uint32_t, f);
  u += 0x7FFFu + ((u >> 16) & 1u);
  return (uint16_t)(u >> 16);
}
static __device__ __forceinline__ uint32_t pack2(float a, float b) {
  return (uint32_t)f2bf(a) | ((uint32_t)f2bf(b) << 16);
}

static __device__ __forceinline__ v8f wmma_bf16(const Frag& a, const Frag& b, v8f acc) {
  return __builtin_amdgcn_wmma_f32_16x16x32_bf16(
      false, a.v, false, b.v, (short)0, acc, false, false);
}

// ---- CDNA5 async Global->LDS DMA (16B per lane), tracked by ASYNCcnt ----
static __device__ __forceinline__ void async_cp_b128(uint32_t* ldsDst, const void* gsrc) {
  uint32_t laddr = (uint32_t)(uintptr_t)(lds_u32*)ldsDst;
  unsigned long long gaddr = (unsigned long long)(uintptr_t)gsrc;
  asm volatile("global_load_async_to_lds_b128 %0, %1, off"
               :: "v"(laddr), "v"(gaddr) : "memory");
}
static __device__ __forceinline__ void wait_async0() {
  asm volatile("s_wait_asynccnt 0x0" ::: "memory");
}

// Convert 16 int4 values (already loaded) into a 32x16 bf16 B-fragment.
static __device__ __forceinline__ Frag conv_bfrag(const int q[16], float s) {
  Frag b;
#pragma unroll
  for (int v = 0; v < 8; ++v)
    b.u[v] = pack2((float)(q[2 * v] - 8) * s, (float)(q[2 * v + 1] - 8) * s);
  return b;
}

// Load a 16x32 bf16 A-fragment from an LDS tile (lds[m][p] = bf16 pair {2p,2p+1})
template <int ROWS>
static __device__ __forceinline__ void load_afrag(Frag& a, const uint32_t lds[ROWS][16],
                                                  int mrow0, int lm, int h) {
#pragma unroll
  for (int v = 0; v < 8; ++v) {
    int p = v + 8 * h;   // contiguous K layout: K = 16h + s (matches B)
    a.u[v] = lds[mrow0 + lm][p];
  }
}

// Stage 8 consecutive bf16 (16B) of one activation row into LDS (async DMA)
template <int ROWS>
static __device__ __forceinline__ void stage_bf16(uint32_t dst[ROWS][16], const uint16_t* arow,
                                                  int n0, int sr, int su, bool svalid) {
  if (svalid) {
    async_cp_b128(&dst[sr][su], arow + n0);
  } else {
    dst[sr][su + 0] = 0u; dst[sr][su + 1] = 0u;
    dst[sr][su + 2] = 0u; dst[sr][su + 3] = 0u;
  }
}

// Stage 16 consecutive bf16 (32B) of one activation row into LDS (async DMA)
template <int ROWS>
static __device__ __forceinline__ void stage_bf16x2(uint32_t dst[ROWS][16], const uint16_t* arow,
                                                    int n0, int sr, int su, bool svalid) {
  if (svalid) {
    async_cp_b128(&dst[sr][su], arow + n0);
    async_cp_b128(&dst[sr][su + 4], arow + n0 + 8);
  } else {
#pragma unroll
    for (int q = 0; q < 8; ++q) dst[sr][su + q] = 0u;
  }
}

// ------------------------------------------------------------------ init
__global__ void init_kernel(int* cnt) {
  if (threadIdx.x < E_EXP) cnt[threadIdx.x] = 0;
}

// ------------------------------------------------------------------ x -> bf16 (once)
__global__ void xconv_kernel(const float* __restrict__ x, uint32_t* __restrict__ xb) {
  int i = blockIdx.x * blockDim.x + threadIdx.x;   // over T*K/2 pairs
  if (i < T_TOK * K_DIM / 2) {
    float2 f = ((const float2*)x)[i];
    xb[i] = pack2(f.x, f.y);
  }
}

// ------------------------------------------------------------------ gating (top-2, renorm)
__global__ void gate_kernel(const float* __restrict__ gating, int* cnt,
                            int* asg_e, int* asg_p, int* tok, float* wgt) {
  int t = blockIdx.x * blockDim.x + threadIdx.x;
  if (t >= T_TOK) return;
  float v[E_EXP];
#pragma unroll
  for (int i = 0; i < E_EXP; ++i) v[i] = gating[t * E_EXP + i];
  int i0 = 0; float b0 = v[0];
#pragma unroll
  for (int i = 1; i < E_EXP; ++i) if (v[i] > b0) { b0 = v[i]; i0 = i; }
  int i1 = (i0 == 0) ? 1 : 0; float b1 = v[i1];
#pragma unroll
  for (int i = 0; i < E_EXP; ++i) if (i != i0 && v[i] > b1) { b1 = v[i]; i1 = i; }
  float e1 = __expf(b1 - b0);
  float w0 = 1.0f / (1.0f + e1);
  float w1 = e1 * w0;
  int p0 = atomicAdd(&cnt[i0], 1);
  int p1 = atomicAdd(&cnt[i1], 1);
  tok[i0 * T_TOK + p0] = t;  wgt[i0 * T_TOK + p0] = w0;
  tok[i1 * T_TOK + p1] = t;  wgt[i1 * T_TOK + p1] = w1;
  asg_e[t * TOPK + 0] = i0;  asg_p[t * TOPK + 0] = p0;
  asg_e[t * TOPK + 1] = i1;  asg_p[t * TOPK + 1] = p1;
}

// ------------------------------------------------------------------ exclusive scan over 8
__global__ void off_kernel(const int* cnt, int* off) {
  if (threadIdx.x == 0 && blockIdx.x == 0) {
    int a = 0;
    for (int e = 0; e < E_EXP; ++e) { off[e] = a; a += cnt[e]; }
  }
}

// ------------------------------------------------------------------ GEMM1 + SwiGLU (weight-resident in LDS)
// Block = (expert, 64 output cols). Phase 1: dequant full K x (gate+up) weight slab
// into LDS once (~257 KB of the 320 KB/WGP). Phase 2: loop over ALL row tiles of the
// expert; activations staged by async Global->LDS DMA overlapping the WMMAs.
__global__ __launch_bounds__(256)
void gemm1_kernel(const uint16_t* __restrict__ xb, const int* __restrict__ qw1,
                  const float* __restrict__ sc1, const int* __restrict__ cnt,
                  const int* __restrict__ off, const int* __restrict__ tok,
                  uint16_t* __restrict__ A /* bf16 [4096][N_DIM] */) {
  const int e = blockIdx.z;
  const int rows = cnt[e];
  if (rows == 0) return;                       // uniform, before any barrier
  const int c0 = blockIdx.x * CT1;

  __shared__ uint32_t wgl[CT1][WP];            // gate weights, bf16 pairs over k
  __shared__ uint32_t wul[CT1][WP];            // up weights
  __shared__ uint32_t ldsA[2][MT][16];         // double-buffered activation tile

  const int tid = threadIdx.x;

  // ---- phase 1: dequantize the weight slab into LDS (coalesced over cols) ----
  const long qbase = (long)e * K_DIM * N2;
  const long sbase = (long)e * (K_DIM / GRP) * N2;
  for (int idx = tid; idx < CT1 * (K_DIM / 2); idx += 256) {
    const int col = idx & (CT1 - 1);           // fast dim -> coalesced
    const int kp  = idx >> 6;                  // k-pair index 0..511
    const int k   = kp * 2;
    const int g   = k >> 7;
    const int cg  = c0 + col;
    const float sG = sc1[sbase + (long)g * N2 + cg];
    const float sU = sc1[sbase + (long)g * N2 + cg + N_DIM];
    const int qg0 = qw1[qbase + (long)k * N2 + cg];
    const int qg1 = qw1[qbase + (long)(k + 1) * N2 + cg];
    const int qu0 = qw1[qbase + (long)k * N2 + cg + N_DIM];
    const int qu1 = qw1[qbase + (long)(k + 1) * N2 + cg + N_DIM];
    wgl[col][kp] = pack2((float)(qg0 - 8) * sG, (float)(qg1 - 8) * sG);
    wul[col][kp] = pack2((float)(qu0 - 8) * sU, (float)(qu1 - 8) * sU);
  }
  __syncthreads();

  // ---- phase 2: loop over row tiles ----
  const int lane = tid & 31, wave = tid >> 5;
  const int h = lane >> 4, lm = lane & 15;
  const int wc = wave & 3;                     // col subtile 0..3
  const int wr = wave >> 2;                    // row half 0..1
  const int bcol = wc * 16 + lm;               // col within LDS weight slab
  const int colw = c0 + bcol;                  // global output col

  const int sr = tid >> 2;                     // staging row 0..63
  const int su = (tid & 3) * 4;                // uint index within 32-k slice
  const int sx = (tid & 3) * 8;                // element offset within slice
  const int rowbase_e = off[e];

  for (int r0 = 0; r0 < rows; r0 += MT) {
    const bool svalid = (r0 + sr) < rows;
    const int st = svalid ? tok[e * T_TOK + r0 + sr] : 0;
    const uint16_t* xrow = xb + (long)st * K_DIM + sx;

    v8f ag[2], au[2];
#pragma unroll
    for (int ms = 0; ms < 2; ++ms) { ag[ms] = (v8f){}; au[ms] = (v8f){}; }

    stage_bf16<MT>(ldsA[0], xrow, 0, sr, su, svalid);
    wait_async0();
    __syncthreads();

    const int NIT = K_DIM / 32;
    for (int it = 0; it < NIT; ++it) {
      const int k0 = it * 32;
      const int cur = it & 1;

      Frag a[2];
      load_afrag<MT>(a[0], ldsA[cur], wr * 32 + 0,  lm, h);
      load_afrag<MT>(a[1], ldsA[cur], wr * 32 + 16, lm, h);

      if (it + 1 < NIT) stage_bf16<MT>(ldsA[cur ^ 1], xrow, k0 + 32, sr, su, svalid);

      Frag bg, bu;
      const int kp0 = (k0 >> 1) + 8 * h;       // B layout: K = k0 + 2v + 16h
#pragma unroll
      for (int v = 0; v < 8; ++v) { bg.u[v] = wgl[bcol][kp0 + v]; bu.u[v] = wul[bcol][kp0 + v]; }

      ag[0] = wmma_bf16(a[0], bg, ag[0]);
      ag[1] = wmma_bf16(a[1], bg, ag[1]);
      au[0] = wmma_bf16(a[0], bu, au[0]);
      au[1] = wmma_bf16(a[1], bu, au[1]);

      wait_async0();       // async DMA of next tile done (overlapped with WMMAs above)
      __syncthreads();     // reads of buf[cur] complete before next iter overwrites it
    }

    // SwiGLU epilogue: silu(gate) * up -> bf16 activation rows
#pragma unroll
    for (int ms = 0; ms < 2; ++ms) {
#pragma unroll
      for (int r = 0; r < 8; ++r) {
        const int M = wr * 32 + ms * 16 + r + 8 * h;   // C/D layout: M = r + 8*half
        if (r0 + M < rows) {
          const float gv = ag[ms][r], uv = au[ms][r];
          A[(long)(rowbase_e + r0 + M) * N_DIM + colw] =
              f2bf(gv / (1.f + __expf(-gv)) * uv);
        }
      }
    }
    __syncthreads();    // before restaging ldsA[0] for the next row tile
  }
}

// ------------------------------------------------------------------ GEMM2 (N -> K), scaled by routing weight
// block: 256 thr (8 waves); tile: 128 rows x 128 k-cols; wave owns 16 cols.
// A-frags on demand; activation tile staged by async Global->LDS DMA.
__global__ __launch_bounds__(256)
void gemm2_kernel(const uint16_t* __restrict__ A /* bf16 [4096][N_DIM] */,
                  const int* __restrict__ qw2, const float* __restrict__ sc2,
                  const int* __restrict__ cnt, const int* __restrict__ off,
                  const float* __restrict__ wgt,
                  float* __restrict__ yb /* [4096][K_DIM] */) {
  const int e = blockIdx.z;
  const int rows = cnt[e];
  const int r0 = blockIdx.y * MT2;
  if (r0 >= rows) return;
  const int kc0 = blockIdx.x * 128;

  __shared__ uint32_t ldsA[2][MT2][16];
  const int tid = threadIdx.x;
  const int lane = tid & 31, wave = tid >> 5;
  const int h = lane >> 4, lm = lane & 15;
  const int rowbase = off[e] + r0;

  const int sr = tid >> 1;                     // staging row 0..127
  const int su = (tid & 1) * 8;                // uint index
  const bool svalid = (r0 + sr) < rows;
  const uint16_t* arowp = A + (long)(rowbase + sr) * N_DIM + (tid & 1) * 16;

  const int col0 = kc0 + wave * 16 + lm;
  const long qcb = (long)e * N_DIM * K_DIM + col0;
  const long scb = (long)e * (N_DIM / GRP) * K_DIM;

  v8f acc[8];
#pragma unroll
  for (int ms = 0; ms < 8; ++ms) acc[ms] = (v8f){};

  stage_bf16x2<MT2>(ldsA[0], arowp, 0, sr, su, svalid);
  wait_async0();
  __syncthreads();

  const int NIT = N_DIM / 32;
  for (int it = 0; it < NIT; ++it) {
    const int n0 = it * 32;
    const int cur = it & 1;

    if (it + 1 < NIT) stage_bf16x2<MT2>(ldsA[cur ^ 1], arowp, n0 + 32, sr, su, svalid);

    if (n0 + 32 < N_DIM)
      __builtin_prefetch(qw2 + qcb + (long)(n0 + 32) * K_DIM, 0, 1);

    const int g = n0 >> 7;
    const float s = sc2[scb + (long)g * K_DIM + col0];

    int q[16];
#pragma unroll
    for (int v = 0; v < 8; ++v) {
      const long n = (long)(n0 + 2 * v + 16 * h);
      q[2 * v]     = qw2[qcb + n * K_DIM];
      q[2 * v + 1] = qw2[qcb + (n + 1) * K_DIM];
    }
    const Frag b = conv_bfrag(q, s);

#pragma unroll
    for (int ms = 0; ms < 8; ++ms) {
      Frag a;
      load_afrag<MT2>(a, ldsA[cur], ms * 16, lm, h);
      acc[ms] = wmma_bf16(a, b, acc[ms]);
    }

    wait_async0();
    __syncthreads();
  }

#pragma unroll
  for (int ms = 0; ms < 8; ++ms) {
#pragma unroll
    for (int r = 0; r < 8; ++r) {
      const int M = ms * 16 + r + 8 * h;
      if (r0 + M < rows) {
        const float w = wgt[e * T_TOK + r0 + M];
        yb[(long)(rowbase + M) * K_DIM + col0] = acc[ms][r] * w;
      }
    }
  }
}

// ------------------------------------------------------------------ deterministic 2-term combine
__global__ void combine_kernel(const float* __restrict__ yb, const int* __restrict__ off,
                               const int* __restrict__ asg_e, const int* __restrict__ asg_p,
                               float* __restrict__ out) {
  int idx = blockIdx.x * blockDim.x + threadIdx.x;
  if (idx >= T_TOK * K_DIM) return;
  const int t = idx >> 10;           // / K_DIM
  const int k = idx & (K_DIM - 1);
  float acc = 0.f;
#pragma unroll
  for (int j = 0; j < TOPK; ++j) {
    const int e = asg_e[t * TOPK + j];
    const int row = off[e] + asg_p[t * TOPK + j];
    acc += yb[(long)row * K_DIM + k];
  }
  out[idx] = acc;
}

// ------------------------------------------------------------------ launch
extern "C" void kernel_launch(void* const* d_in, const int* in_sizes, int n_in,
                              void* d_out, int out_size, void* d_ws, size_t ws_size,
                              hipStream_t stream) {
  (void)in_sizes; (void)n_in; (void)out_size; (void)ws_size;
  const float* x      = (const float*)d_in[0];
  const float* gating = (const float*)d_in[1];
  const int*   qw1    = (const int*)d_in[2];
  const float* sc1    = (const float*)d_in[3];
  // d_in[4] g_idx1: trivial k//128 mapping, folded into indexing
  const int*   qw2    = (const int*)d_in[5];
  const float* sc2    = (const float*)d_in[6];
  // d_in[7] g_idx2, d_in[8] topk (=2): folded

  char* ws = (char*)d_ws;
  int*      cnt   = (int*)(ws + 0);
  int*      off   = (int*)(ws + 32);
  int*      asg_e = (int*)(ws + 64);
  int*      asg_p = (int*)(ws + 16448);
  int*      tok   = (int*)(ws + 32832);
  float*    wgt   = (float*)(ws + 98368);
  uint32_t* xbp   = (uint32_t*)(ws + 163904);                  // bf16 pairs [T*K/2]
  uint16_t* Abuf  = (uint16_t*)(ws + 4358208);                 // bf16 [4096][N_DIM]
  float*    yb    = (float*)(ws + 27426880);                   // f32 [4096][K_DIM]

  hipLaunchKernelGGL(init_kernel, dim3(1), dim3(32), 0, stream, cnt);
  hipLaunchKernelGGL(xconv_kernel, dim3((T_TOK * K_DIM / 2 + 255) / 256), dim3(256), 0, stream,
                     x, xbp);
  hipLaunchKernelGGL(gate_kernel, dim3(T_TOK / 256), dim3(256), 0, stream,
                     gating, cnt, asg_e, asg_p, tok, wgt);
  hipLaunchKernelGGL(off_kernel, dim3(1), dim3(1), 0, stream, cnt, off);
  hipLaunchKernelGGL(gemm1_kernel, dim3(N_DIM / CT1, 1, E_EXP), dim3(256), 0, stream,
                     (const uint16_t*)xbp, qw1, sc1, cnt, off, tok, Abuf);
  hipLaunchKernelGGL(gemm2_kernel, dim3(K_DIM / 128, T_TOK / MT2, E_EXP), dim3(256), 0, stream,
                     Abuf, qw2, sc2, cnt, off, wgt, yb);
  hipLaunchKernelGGL(combine_kernel, dim3((T_TOK * K_DIM + 255) / 256), dim3(256), 0, stream,
                     yb, off, asg_e, asg_p, (float*)d_out);
}
